// MultiHeadSelfAttention_1906965480105
// MI455X (gfx1250) — matrix operand order repaired and checked by reading, not compiled, not run on hardware
//
#include <hip/hip_runtime.h>

// ---------------------------------------------------------------------------
// MI455X (gfx1250) multi-head causal self-attention, f16 WMMA pipeline.
//   X[B,S,D] --W_Q/W_K/W_V--> Q,K[B*NH,S,64], V^T[B*NH,64,S]
//   flash attention (online softmax, causal) --> O[B,S,1024] (f16)
//   O @ H + b_h --> out[B,S,D] (f32)
// All matmuls: v_wmma_f32_16x16x32_f16. All fragment loads: 2x ds_load_b128.
// Tile staging uses global_load_async_to_lds_b128 (ASYNCcnt) when the
// toolchain exposes the builtins, else falls back to uint4 copies.
// ---------------------------------------------------------------------------

typedef __attribute__((ext_vector_type(16))) _Float16 v16h;
typedef __attribute__((ext_vector_type(8)))  float    v8f;
typedef int v4i_vs __attribute__((vector_size(16)));  // matches builtin param

namespace {
constexpr int Bc  = 2;
constexpr int Sc  = 2048;
constexpr int Dc  = 1024;
constexpr int NHc = 16;
constexpr int DKc = 64;
constexpr int DVc = 64;
constexpr int DH  = NHc * DVc;  // 1024
}

#if defined(__gfx1250__) && \
    __has_builtin(__builtin_amdgcn_global_load_async_to_lds_b128) && \
    __has_builtin(__builtin_amdgcn_s_wait_asynccnt)
#define USE_ASYNC_LDS 1
#else
#define USE_ASYNC_LDS 0
#endif

// Stage 16 bytes global -> LDS (async DMA path when available).
__device__ __forceinline__ void stage_b128(_Float16* dst_lds,
                                           const _Float16* src_g) {
#if USE_ASYNC_LDS
  __builtin_amdgcn_global_load_async_to_lds_b128(
      (__attribute__((address_space(1))) v4i_vs*)src_g,
      (__attribute__((address_space(3))) v4i_vs*)dst_lds,
      /*imm offset*/ 0, /*cpol*/ 0);
#else
  *(uint4*)dst_lds = *(const uint4*)src_g;
#endif
}

__device__ __forceinline__ void stage_wait() {
#if USE_ASYNC_LDS
  __builtin_amdgcn_s_wait_asynccnt(0);
#endif
}

union Frag16 { uint4 q[2]; v16h v; };

// A-matrix 16x32 f16 (ISA 7.12.2): lane L -> row m=L%16; halfs h[0..7] are
// K = (L/16)*8 .. +7, h[8..15] are K = 16+(L/16)*8 .. +7  ==> 2x b128.
__device__ __forceinline__ v16h load_A_frag(const _Float16* lds, int stride) {
  const int lane = threadIdx.x & 31;
  const int m = lane & 15, hi = lane >> 4;
  Frag16 a;
  a.q[0] = *(const uint4*)&lds[m * stride + hi * 8];
  a.q[1] = *(const uint4*)&lds[m * stride + 16 + hi * 8];
  return a.v;
}

// B-matrix 32x16 f16 from an LDS tile stored [n][k] (K-contiguous rows,
// row stride `stride`): lane L -> col n=L%16; h[0..7] = K=(L/16)*8..+7,
// h[8..15] = K=16+(L/16)*8..+7 at fixed n  ==> 2x b128.
__device__ __forceinline__ v16h load_Bn_frag(const _Float16* lds, int stride) {
  const int lane = threadIdx.x & 31;
  const int n = lane & 15, hi = lane >> 4;
  const _Float16* p = &lds[n * stride + hi * 8];
  Frag16 b;
  b.q[0] = *(const uint4*)p;
  b.q[1] = *(const uint4*)(p + 16);
  return b.v;
}

// --- conversions ------------------------------------------------------------
__global__ void cvt_f16_kernel(const float* __restrict__ in,
                               _Float16* __restrict__ out, int n) {
  int i = blockIdx.x * blockDim.x + threadIdx.x;
  if (i < n) out[i] = (_Float16)in[i];
}

// W [NH, D, 64] f32 -> Wt [NH, 64, D] f16 (K-contiguous rows for B tiles)
__global__ void cvt_w_tr_kernel(const float* __restrict__ in,
                                _Float16* __restrict__ out) {
  int i = blockIdx.x * blockDim.x + threadIdx.x;
  if (i < NHc * Dc * DKc) {
    const int h = i / (Dc * DKc);
    const int rem = i - h * (Dc * DKc);
    const int d = rem / DKc, n = rem % DKc;
    out[((size_t)h * DKc + n) * Dc + d] = (_Float16)in[i];
  }
}

// H [DH, D] f32 -> Ht [D, DH] f16
__global__ void cvt_h_tr_kernel(const float* __restrict__ in,
                                _Float16* __restrict__ out) {
  int i = blockIdx.x * blockDim.x + threadIdx.x;
  if (i < DH * Dc) {
    const int c = i / Dc, d = i - c * Dc;
    out[(size_t)d * DH + c] = (_Float16)in[i];
  }
}

// --- QKV projection: per (b,h) GEMM [S x D] * [D x 64] ----------------------
// 4 waves/block -> 64x64 block tile (wave w owns rows w*16..+15).
// B tile (Wt, K-contiguous) staged once per K-step and shared by all waves.
template <bool TRANSPOSED>
__global__ __launch_bounds__(128) void qkv_gemm_kernel(
    const _Float16* __restrict__ Xh,   // [B, S, D]
    const _Float16* __restrict__ Wt,   // [NH, 64, D]
    _Float16* __restrict__ Out) {      // [B*NH, S, 64] or [B*NH, 64, S]
  __shared__ __align__(16) _Float16 As[4 * 16 * 32];  // per-wave A tiles
  __shared__ __align__(16) _Float16 Bs[64 * 32];      // [n][k]
  const int qt = blockIdx.x, h = blockIdx.y, b = blockIdx.z;
  const int t = threadIdx.x, lane = t & 31, w = t >> 5;
  const int bh = b * NHc + h;
  const _Float16* Xp = Xh + (size_t)(b * Sc + qt * 64 + w * 16) * Dc;
  const _Float16* Wp = Wt + (size_t)h * DKc * Dc;
  _Float16* Aw = As + w * 512;

  v8f acc[4] = {};
  for (int d0 = 0; d0 < Dc; d0 += 32) {
    __syncthreads();
#pragma unroll
    for (int i = t; i < 256; i += 128) {  // B tile: 64 rows(n) x 32 (k)
      const int row = i >> 2, c = (i & 3) * 8;
      stage_b128(&Bs[row * 32 + c], &Wp[(size_t)row * Dc + d0 + c]);
    }
#pragma unroll
    for (int i = lane; i < 64; i += 32) {  // A tile: 16 x 32 per wave
      const int r = i >> 2, c = (i & 3) * 8;
      stage_b128(&Aw[r * 32 + c], &Xp[(size_t)r * Dc + d0 + c]);
    }
    stage_wait();
    __syncthreads();
    const v16h af = load_A_frag(Aw, 32);
#pragma unroll
    for (int nc = 0; nc < 4; ++nc) {
      const v16h bf = load_Bn_frag(Bs + nc * 16 * 32, 32);
      acc[nc] = __builtin_amdgcn_wmma_f32_16x16x32_f16(
          false, af, false, bf, (short)0, acc[nc], false, false);
    }
  }
  const int n = lane & 15, hi = lane >> 4;
#pragma unroll
  for (int nc = 0; nc < 4; ++nc)
#pragma unroll
    for (int r = 0; r < 8; ++r) {
      const int m = w * 16 + r + 8 * hi;
      const _Float16 val = (_Float16)acc[nc][r];
      if constexpr (TRANSPOSED)   // V^T: [B*NH, 64(dv), S]
        Out[((size_t)bh * DKc + nc * 16 + n) * Sc + qt * 64 + m] = val;
      else                        // Q/K: [B*NH, S, 64]
        Out[((size_t)bh * Sc + qt * 64 + m) * DKc + nc * 16 + n] = val;
    }
}

// --- Flash attention (causal): 4 waves/block = 64 queries ------------------
__global__ __launch_bounds__(128) void attn_kernel(
    const _Float16* __restrict__ Q,    // [B*NH, S, 64]
    const _Float16* __restrict__ Km,   // [B*NH, S, 64]
    const _Float16* __restrict__ Vt,   // [B*NH, 64, S]  (transposed V)
    _Float16* __restrict__ Oh) {       // [B, S, NH*64]
  __shared__ __align__(16) _Float16 Qs[64 * 64];   // 8 KB, staged once
  __shared__ __align__(16) _Float16 Ks[32 * 64];   // [key][dk] == [n][k]
  __shared__ __align__(16) _Float16 Vts[64 * 32];  // [dv][key] == [n][k]
  __shared__ __align__(16) _Float16 Ps[4 * 16 * 32];
  const int qt = blockIdx.x, h = blockIdx.y, b = blockIdx.z;
  const int t = threadIdx.x, lane = t & 31, w = t >> 5;
  const int n = lane & 15, hi = lane >> 4;
  const int bh = b * NHc + h;
  const float NEG_INF = -__builtin_huge_valf();

  const _Float16* Qp  = Q  + ((size_t)bh * Sc + qt * 64) * DKc;
  const _Float16* Kb  = Km + (size_t)bh * Sc * DKc;
  const _Float16* Vtp = Vt + (size_t)bh * DKc * Sc;
  _Float16* Pw = Ps + w * 512;

#pragma unroll
  for (int i = t; i < 512; i += 128)
    stage_b128(&Qs[i * 8], &Qp[i * 8]);
  stage_wait();
  __syncthreads();
  const v16h aq0 = load_A_frag(Qs + w * 16 * 64, 64);
  const v16h aq1 = load_A_frag(Qs + w * 16 * 64 + 32, 64);

  float mrow[8], lrow[8];
#pragma unroll
  for (int r = 0; r < 8; ++r) { mrow[r] = NEG_INF; lrow[r] = 0.0f; }
  v8f o[4] = {};

  const int njt = 2 * qt + 2;  // 32-key tiles up to block causal bound
  for (int j = 0; j < njt; ++j) {
    const _Float16* Kp = Kb + (size_t)j * 32 * DKc;
    __syncthreads();
#pragma unroll
    for (int i = t; i < 256; i += 128)  // K tile 32x64 (contiguous)
      stage_b128(&Ks[i * 8], &Kp[i * 8]);
#pragma unroll
    for (int i = t; i < 256; i += 128) {  // V^T tile 64(dv) x 32(key)
      const int row = i >> 2, c = (i & 3) * 8;
      stage_b128(&Vts[row * 32 + c], &Vtp[(size_t)row * Sc + j * 32 + c]);
    }
    if (j + 1 < njt)  // global_prefetch_b8 next K tile
      __builtin_prefetch(Kp + 32 * DKc, 0, 1);
    stage_wait();
    __syncthreads();

    // Scores S = Q.K^T * 1/8, two 16-key halves, DK split 32+32.
    v8f sc[2];
#pragma unroll
    for (int c16 = 0; c16 < 2; ++c16) {
      v8f s = {};
      s = __builtin_amdgcn_wmma_f32_16x16x32_f16(
          false, aq0, false, load_Bn_frag(Ks + c16 * 16 * 64, 64),
          (short)0, s, false, false);
      s = __builtin_amdgcn_wmma_f32_16x16x32_f16(
          false, aq1, false, load_Bn_frag(Ks + c16 * 16 * 64 + 32, 64),
          (short)0, s, false, false);
      sc[c16] = s;
    }

    // Scale + causal mask + row max (reduce across the 16 lanes of a half).
    float rmax[8];
#pragma unroll
    for (int r = 0; r < 8; ++r) {
      const int q = qt * 64 + w * 16 + r + 8 * hi;
#pragma unroll
      for (int c16 = 0; c16 < 2; ++c16) {
        const int key = j * 32 + c16 * 16 + n;
        float vv = sc[c16][r] * 0.125f;
        if (key > q) vv = NEG_INF;
        sc[c16][r] = vv;
      }
      float rm = fmaxf(sc[0][r], sc[1][r]);
#pragma unroll
      for (int off = 8; off >= 1; off >>= 1)
        rm = fmaxf(rm, __shfl_xor(rm, off, 32));
      rmax[r] = rm;
    }

    // Online softmax update (fully-masked tiles: rmax=-inf -> p=0, alpha=1).
    float alpha[8];
#pragma unroll
    for (int r = 0; r < 8; ++r) {
      const float mnew = fmaxf(mrow[r], rmax[r]);
      alpha[r] = __expf(mrow[r] - mnew);
      mrow[r] = mnew;
      const float p0 = __expf(sc[0][r] - mnew);
      const float p1 = __expf(sc[1][r] - mnew);
      sc[0][r] = p0; sc[1][r] = p1;
      float rs = p0 + p1;
#pragma unroll
      for (int off = 8; off >= 1; off >>= 1)
        rs += __shfl_xor(rs, off, 32);
      lrow[r] = lrow[r] * alpha[r] + rs;
    }
#pragma unroll
    for (int nc = 0; nc < 4; ++nc)
#pragma unroll
      for (int r = 0; r < 8; ++r) o[nc][r] *= alpha[r];

    // P (f32, C layout) -> per-wave LDS 16x32 f16 -> A fragment.
#pragma unroll
    for (int r = 0; r < 8; ++r) {
      Pw[(r + 8 * hi) * 32 + n]      = (_Float16)sc[0][r];
      Pw[(r + 8 * hi) * 32 + 16 + n] = (_Float16)sc[1][r];
    }
    __syncthreads();
    const v16h aP = load_A_frag(Pw, 32);

    // O += P.V   (V^T tile is [dv][key] = [n][k], K-contiguous)
#pragma unroll
    for (int nc = 0; nc < 4; ++nc) {
      o[nc] = __builtin_amdgcn_wmma_f32_16x16x32_f16(
          false, aP, false, load_Bn_frag(Vts + nc * 16 * 32, 32),
          (short)0, o[nc], false, false);
    }
  }

  // Normalize, store O f16 in [B, S, NH*64] (heads concatenated).
#pragma unroll
  for (int nc = 0; nc < 4; ++nc)
#pragma unroll
    for (int r = 0; r < 8; ++r) {
      const int m = w * 16 + r + 8 * hi;
      Oh[((size_t)b * Sc + qt * 64 + m) * DH + h * DVc + nc * 16 + n] =
          (_Float16)(o[nc][r] / lrow[r]);
    }
}

// --- Output projection: [B*S x 1024] * H + bias, f32 out --------------------
__global__ __launch_bounds__(128) void proj_kernel(
    const _Float16* __restrict__ Oh,   // [B*S, 1024]
    const _Float16* __restrict__ Ht,   // [D, DH] (H^T, K-contiguous rows)
    const float* __restrict__ bias,    // [D]
    float* __restrict__ Out) {         // [B*S, D]
  __shared__ __align__(16) _Float16 As[4 * 16 * 32];
  __shared__ __align__(16) _Float16 Bs[64 * 32];
  const int mt = blockIdx.x, nt = blockIdx.y;
  const int t = threadIdx.x, lane = t & 31, w = t >> 5;
  const _Float16* Ap = Oh + (size_t)(mt * 64 + w * 16) * DH;
  const _Float16* Hp = Ht + (size_t)nt * 64 * Dc;  // rows = output cols
  _Float16* Aw = As + w * 512;

  v8f acc[4] = {};
  for (int c0 = 0; c0 < DH; c0 += 32) {
    __syncthreads();
#pragma unroll
    for (int i = t; i < 256; i += 128) {
      const int row = i >> 2, c = (i & 3) * 8;
      stage_b128(&Bs[row * 32 + c], &Hp[(size_t)row * Dc + c0 + c]);
    }
#pragma unroll
    for (int i = lane; i < 64; i += 32) {
      const int r = i >> 2, c = (i & 3) * 8;
      stage_b128(&Aw[r * 32 + c], &Ap[(size_t)r * DH + c0 + c]);
    }
    stage_wait();
    __syncthreads();
    const v16h af = load_A_frag(Aw, 32);
#pragma unroll
    for (int nc = 0; nc < 4; ++nc) {
      const v16h bf = load_Bn_frag(Bs + nc * 16 * 32, 32);
      acc[nc] = __builtin_amdgcn_wmma_f32_16x16x32_f16(
          false, af, false, bf, (short)0, acc[nc], false, false);
    }
  }
  const int n = lane & 15, hi = lane >> 4;
#pragma unroll
  for (int nc = 0; nc < 4; ++nc)
#pragma unroll
    for (int r = 0; r < 8; ++r) {
      const int m = mt * 64 + w * 16 + r + 8 * hi;
      const int col = nt * 64 + nc * 16 + n;
      Out[(size_t)m * Dc + col] = acc[nc][r] + bias[col];
    }
}

// ---------------------------------------------------------------------------
extern "C" void kernel_launch(void* const* d_in, const int* in_sizes, int n_in,
                              void* d_out, int out_size, void* d_ws,
                              size_t ws_size, hipStream_t stream) {
  (void)in_sizes; (void)n_in; (void)out_size; (void)ws_size;
  const float* X   = (const float*)d_in[0];
  const float* W_Q = (const float*)d_in[1];
  const float* W_K = (const float*)d_in[2];
  const float* W_V = (const float*)d_in[3];
  const float* Hm  = (const float*)d_in[4];
  const float* b_h = (const float*)d_in[5];
  float* out = (float*)d_out;

  // Workspace carve-up (all f16). Total = 48 MB.
  char* ws = (char*)d_ws;
  const size_t nX = (size_t)Bc * Sc * Dc;          // 4M elems
  const size_t nW = (size_t)NHc * Dc * DKc;        // 1M elems
  const size_t nQ = (size_t)Bc * NHc * Sc * DKc;   // 4M elems
  _Float16* Xh  = (_Float16*)ws;  ws += nX * 2;
  _Float16* Wqt = (_Float16*)ws;  ws += nW * 2;    // [NH,64,D]
  _Float16* Wkt = (_Float16*)ws;  ws += nW * 2;
  _Float16* Wvt = (_Float16*)ws;  ws += nW * 2;
  _Float16* Hht = (_Float16*)ws;  ws += nW * 2;    // [D,DH]
  _Float16* Qb  = (_Float16*)ws;  ws += nQ * 2;    // [B*NH,S,64]
  _Float16* Kb  = (_Float16*)ws;  ws += nQ * 2;    // [B*NH,S,64]
  _Float16* Vtb = (_Float16*)ws;  ws += nQ * 2;    // [B*NH,64,S]
  _Float16* Ob  = (_Float16*)ws;  ws += nX * 2;    // [B,S,DH]

  const int T = 256;
  cvt_f16_kernel <<<(int)((nX + T - 1) / T), T, 0, stream>>>(X, Xh, (int)nX);
  cvt_w_tr_kernel<<<(int)((nW + T - 1) / T), T, 0, stream>>>(W_Q, Wqt);
  cvt_w_tr_kernel<<<(int)((nW + T - 1) / T), T, 0, stream>>>(W_K, Wkt);
  cvt_w_tr_kernel<<<(int)((nW + T - 1) / T), T, 0, stream>>>(W_V, Wvt);
  cvt_h_tr_kernel<<<(int)((nW + T - 1) / T), T, 0, stream>>>(Hm, Hht);

  dim3 gqkv(Sc / 64, NHc, Bc);
  qkv_gemm_kernel<false><<<gqkv, 128, 0, stream>>>(Xh, Wqt, Qb);
  qkv_gemm_kernel<false><<<gqkv, 128, 0, stream>>>(Xh, Wkt, Kb);
  qkv_gemm_kernel<true ><<<gqkv, 128, 0, stream>>>(Xh, Wvt, Vtb);

  attn_kernel<<<gqkv, 128, 0, stream>>>(Qb, Kb, Vtb, Ob);

  dim3 gproj(Bc * Sc / 64, Dc / 64);
  proj_kernel<<<gproj, 128, 0, stream>>>(Ob, Hht, b_h, out);
}
